// GCN_32495722561552
// MI455X (gfx1250) — matrix operand order repaired and checked by reading, not compiled, not run on hardware
//
#include <hip/hip_runtime.h>
#include <hip/hip_bf16.h>

// ---------------------------------------------------------------------------
// GCN (4 layers) + feature-correlation metrics for MI455X (gfx1250, wave32).
//   h = x
//   for l in 0..3: z = h @ W_l ; h = segment_sum(norm * z[col], row)
//                  corr metric after layers 2 and 3 (pre-ReLU); ReLU for l<3
// Dense GEMMs and Gram matrices use V_WMMA_F32_16X16X32_F16.
// Sparse aggregation uses native GLOBAL_ATOMIC_ADD_F32 (inline asm) --
// z (51MB) + hout (51MB) both fit in the 192MB L2, so the edge phase runs at
// L2 bandwidth, which is the roofline floor for this graph workload.
// ---------------------------------------------------------------------------

typedef __attribute__((ext_vector_type(16))) _Float16 v16h;
typedef __attribute__((ext_vector_type(8)))  _Float16 v8h;
typedef __attribute__((ext_vector_type(8)))  float    v8f;

union V16 { v16h v; v8h h[2]; };

#define NN   100000
#define EE   1600000
#define FIN  256
#define HH   128
#define CC   64

// Native f32 atomic add (no return -> STOREcnt; s_endpgm implicit wait-idle
// drains it). Guarantees GLOBAL_ATOMIC_ADD_F32 instead of a CAS loop.
__device__ __forceinline__ void atomic_add_f32(float* p, float v) {
  asm volatile("global_atomic_add_f32 %0, %1, off" :: "v"(p), "v"(v) : "memory");
}

// ---------------------------------------------------------------- utilities
__global__ void zero_u32(unsigned* __restrict__ p, long n) {
  long i = (long)blockIdx.x * blockDim.x + threadIdx.x;
  if (i < n) p[i] = 0u;
}

__global__ void deg_count(const int* __restrict__ rows, unsigned* __restrict__ deg, int e) {
  int i = blockIdx.x * blockDim.x + threadIdx.x;
  if (i < e) atomicAdd(&deg[rows[i]], 1u);
}

__global__ void make_dinv(const unsigned* __restrict__ deg, float* __restrict__ dinv, int n) {
  int i = blockIdx.x * blockDim.x + threadIdx.x;
  if (i < n) dinv[i] = rsqrtf((float)(deg[i] + 1u));   // +1 self loop; deg>0 always
}

__global__ void f32_to_f16(const float* __restrict__ src, _Float16* __restrict__ dst,
                           long n, int relu) {
  long i = (long)blockIdx.x * blockDim.x + threadIdx.x;
  if (i < n) {
    float v = src[i];
    if (relu && v < 0.0f) v = 0.0f;
    dst[i] = (_Float16)v;
  }
}

__global__ void relu_f16(_Float16* __restrict__ h, long n) {
  long i = (long)blockIdx.x * blockDim.x + threadIdx.x;
  if (i < n) { _Float16 v = h[i]; h[i] = (v < (_Float16)0.0f) ? (_Float16)0.0f : v; }
}

// W [din,dout] f32 row-major -> Wt [dout,din] f16 (column-major W) so the WMMA
// B-operand per-lane loads are contiguous 16B vectors.
__global__ void wconv(const float* __restrict__ W, _Float16* __restrict__ Wt,
                      int din, int dout) {
  int idx = blockIdx.x * blockDim.x + threadIdx.x;
  if (idx < din * dout) {
    int i = idx / dout, o = idx - i * dout;
    Wt[(size_t)o * din + i] = (_Float16)W[idx];
  }
}

// ------------------------------------------------------------- dense GEMM
// Z[N,dout] = A[N,din](f16) x Wt[dout,din](f16, pre-transposed).
// One wave per 16x16 output tile; block = (dout/16) waves covering one M-tile.
// Lane layout per CDNA5 ISA 7.12.2 (16-bit A: lanes<16 K in {0..7,16..23}).
__global__ void gemm_wmma(const _Float16* __restrict__ A, const _Float16* __restrict__ Bt,
                          float* __restrict__ C, int din, int dout) {
  const int wave = threadIdx.x >> 5;
  const int lane = threadIdx.x & 31;
  const int half = lane >> 4;
  const int l15  = lane & 15;
  const int mbase = blockIdx.x * 16;
  const int nbase = wave * 16;

  const _Float16* arow = A  + (size_t)(mbase + l15) * din + half * 8;   // A half-select
  const _Float16* bcol = Bt + (size_t)(nbase + l15) * din + half * 8;   // B half-select

  v8f c = {};
  for (int k = 0; k < din; k += 32) {
    V16 a, b;
    a.h[0] = *(const v8h*)(arow + k);        // K = k+hs .. k+hs+7
    a.h[1] = *(const v8h*)(arow + k + 16);   // K = k+hs+16 .. k+hs+23
    b.h[0] = *(const v8h*)(bcol + k);        // K = k+hs .. k+hs+7
    b.h[1] = *(const v8h*)(bcol + k + 16);   // K = k+hs+16 .. k+hs+23
    c = __builtin_amdgcn_wmma_f32_16x16x32_f16(false, a.v, false, b.v,
                                               (short)0, c, false, false);
  }

  float* crow = C + (size_t)mbase * dout + nbase + l15;
#pragma unroll
  for (int i = 0; i < 8; ++i)
    crow[(size_t)(i + half * 8) * dout] = c[i];
}

// ------------------------------------------------------- sparse aggregation
// One wave per edge (E directed edges + N self loops). Per lane: one b128
// (d=128) or b64 (d=64) gather of z[col], then native f32 atomic adds into
// hout[row]. Edge indices read via SMEM (readfirstlane'd wave id).
__global__ void aggregate(const int* __restrict__ rows, const int* __restrict__ cols,
                          const float* __restrict__ dinv, const float* __restrict__ z,
                          float* __restrict__ hout, int d, int e, int etot) {
  int wid  = (int)(((long)blockIdx.x * blockDim.x + threadIdx.x) >> 5);
  wid = __builtin_amdgcn_readfirstlane(wid);       // wave-uniform -> SGPR
  const int lane = threadIdx.x & 31;
  if (wid >= etot) return;
  int r, c;
  if (wid < e) { r = rows[wid]; c = cols[wid]; } else { r = c = wid - e; }
  const float w = dinv[r] * dinv[c];
  if (d == 128) {
    const float4 zv = *(const float4*)(z + (size_t)c * 128 + lane * 4);
    float* hd = hout + (size_t)r * 128 + lane * 4;
    atomic_add_f32(hd + 0, w * zv.x);
    atomic_add_f32(hd + 1, w * zv.y);
    atomic_add_f32(hd + 2, w * zv.z);
    atomic_add_f32(hd + 3, w * zv.w);
  } else {  // d == 64
    const float2 zv = *(const float2*)(z + (size_t)c * 64 + lane * 2);
    float* hd = hout + (size_t)r * 64 + lane * 2;
    atomic_add_f32(hd + 0, w * zv.x);
    atomic_add_f32(hd + 1, w * zv.y);
  }
}

// ----------------------------------------------------------- Gram = H^T H
// Split-K: blockIdx.z selects a chunk of rows; 16x16 G tiles via WMMA,
// f32 atomic accumulate into G.
__global__ void gram_wmma(const _Float16* __restrict__ H, float* __restrict__ G,
                          int d, int rows_per_chunk) {
  const int lane = threadIdx.x & 31;
  const int half = lane >> 4;
  const int l15  = lane & 15;
  const int ci = blockIdx.x * 16 + l15;   // A column (G tile row index base)
  const int cj = blockIdx.y * 16 + l15;   // B column
  const int kstart = blockIdx.z * rows_per_chunk;

  v8f c = {};
  for (int k0 = 0; k0 < rows_per_chunk; k0 += 32) {
    const int kb = kstart + k0;
    V16 a, b;
#pragma unroll
    for (int i = 0; i < 8; ++i) {
      a.v[i]     = H[(size_t)(kb + half * 8 + i)      * d + ci];
      a.v[i + 8] = H[(size_t)(kb + half * 8 + 16 + i) * d + ci];
    }
#pragma unroll
    for (int i = 0; i < 8; ++i) {
      b.v[i]     = H[(size_t)(kb + half * 8 + i)      * d + cj];
      b.v[i + 8] = H[(size_t)(kb + half * 8 + 16 + i) * d + cj];
    }
    c = __builtin_amdgcn_wmma_f32_16x16x32_f16(false, a.v, false, b.v,
                                               (short)0, c, false, false);
  }
#pragma unroll
  for (int i = 0; i < 8; ++i) {
    int gi = blockIdx.x * 16 + i + half * 8;
    atomic_add_f32(G + (size_t)gi * d + cj, c[i]);
  }
}

// --------------------------------------------------------- column sums of h
__global__ void colsum_k(const float* __restrict__ h, float* __restrict__ cs,
                         int d, int n) {
  int j = threadIdx.x;                       // blockDim.x == d
  float s = 0.0f;
  for (int r = blockIdx.x; r < n; r += gridDim.x) s += h[(size_t)r * d + j];
  atomic_add_f32(&cs[j], s);
}

// -------------------------------------------- corr scalar from G and colsum
__global__ void corr_finalize(const float* __restrict__ G, const float* __restrict__ cs,
                              int d, float n, float* __restrict__ out) {
  __shared__ float mean[HH];
  __shared__ float red[256];
  int t = threadIdx.x;
  for (int j = t; j < d; j += blockDim.x) mean[j] = cs[j] / n;
  __syncthreads();
  float acc = 0.0f;
  for (int idx = t; idx < d * d; idx += blockDim.x) {
    int i = idx / d, j = idx - i * d;
    if (j > i) {
      float cov = G[idx]          - n * mean[i] * mean[j];
      float cii = G[i * d + i]    - n * mean[i] * mean[i];
      float cjj = G[j * d + j]    - n * mean[j] * mean[j];
      float sd  = sqrtf(fmaxf(cii, 1e-12f)) * sqrtf(fmaxf(cjj, 1e-12f));
      acc += fabsf(cov / sd);
    }
  }
  red[t] = acc;
  __syncthreads();
  for (int s = blockDim.x >> 1; s > 0; s >>= 1) {
    if (t < s) red[t] += red[t + s];
    __syncthreads();
  }
  if (t == 0) *out = red[0] / ((float)d * (float)(d - 1) * 0.5f);
}

// ---------------------------------------------------------------------------
extern "C" void kernel_launch(void* const* d_in, const int* in_sizes, int n_in,
                              void* d_out, int out_size, void* d_ws, size_t ws_size,
                              hipStream_t stream) {
  const float* x   = (const float*)d_in[0];
  const int*   ei  = (const int*)d_in[1];
  const int*   rows = ei;
  const int*   cols = ei + EE;
  const float* W[4] = {(const float*)d_in[2], (const float*)d_in[3],
                       (const float*)d_in[4], (const float*)d_in[5]};
  float* outf = (float*)d_out;

  // ---- workspace carve-up (256B aligned slabs) ----
  char* wsb = (char*)d_ws;
  size_t off = 0;
  auto take = [&](size_t bytes) -> void* {
    void* p = wsb + off;
    off += (bytes + 255) & ~(size_t)255;
    return p;
  };
  unsigned* deg  = (unsigned*)take((size_t)NN * 4);
  float*    dinv = (float*)   take((size_t)NN * 4);
  _Float16* h16  = (_Float16*)take((size_t)NN * FIN * 2);   // layer input (f16)
  float*    z    = (float*)   take((size_t)NN * HH * 4);    // GEMM output
  float*    hout = (float*)   take((size_t)NN * HH * 4);    // aggregated output
  _Float16* wt0  = (_Float16*)take((size_t)FIN * HH * 2);
  _Float16* wt1  = (_Float16*)take((size_t)HH * HH * 2);
  _Float16* wt2  = (_Float16*)take((size_t)HH * HH * 2);
  _Float16* wt3  = (_Float16*)take((size_t)HH * CC * 2);
  float*    cs   = (float*)   take((size_t)HH * 4);
  float*    G    = (float*)   take((size_t)HH * HH * 4);

  auto blocks = [](long n, int bs) { return (unsigned)((n + bs - 1) / bs); };

  // ---- normalization: deg (row occurrences + self loop) -> dinv ----
  zero_u32<<<blocks(NN, 256), 256, 0, stream>>>(deg, NN);
  deg_count<<<blocks(EE, 256), 256, 0, stream>>>(rows, deg, EE);
  make_dinv<<<blocks(NN, 256), 256, 0, stream>>>(deg, dinv, NN);

  // ---- weight conversion (transposed f16) ----
  wconv<<<blocks((long)FIN * HH, 256), 256, 0, stream>>>(W[0], wt0, FIN, HH);
  wconv<<<blocks((long)HH * HH, 256), 256, 0, stream>>>(W[1], wt1, HH, HH);
  wconv<<<blocks((long)HH * HH, 256), 256, 0, stream>>>(W[2], wt2, HH, HH);
  wconv<<<blocks((long)HH * CC, 256), 256, 0, stream>>>(W[3], wt3, HH, CC);

  // ---- input features -> f16 ----
  f32_to_f16<<<blocks((long)NN * FIN, 256), 256, 0, stream>>>(x, h16, (long)NN * FIN, 0);

  const _Float16* WT[4] = {wt0, wt1, wt2, wt3};
  const int DIN[4]  = {FIN, HH, HH, HH};
  const int DOUT[4] = {HH, HH, HH, CC};
  const int ETOT = EE + NN;

  for (int l = 0; l < 4; ++l) {
    const int din = DIN[l], dout = DOUT[l];
    float* hdst = (l == 3) ? outf : hout;

    // z = h16 @ W_l   (WMMA; M = 100000 = 6250 exact 16-row tiles)
    gemm_wmma<<<dim3(NN / 16), 32 * (dout / 16), 0, stream>>>(h16, WT[l], z, din, dout);

    // hdst = 0 ; segment-sum of norm-scaled messages
    zero_u32<<<blocks((long)NN * dout, 256), 256, 0, stream>>>((unsigned*)hdst, (long)NN * dout);
    aggregate<<<blocks((long)ETOT * 32, 256), 256, 0, stream>>>(rows, cols, dinv, z,
                                                                hdst, dout, EE, ETOT);

    if (l >= 2) {
      // correlation metric on pre-ReLU h
      f32_to_f16<<<blocks((long)NN * dout, 256), 256, 0, stream>>>(hdst, h16,
                                                                   (long)NN * dout, 0);
      zero_u32<<<blocks(dout, 256), 256, 0, stream>>>((unsigned*)cs, dout);
      zero_u32<<<blocks((long)dout * dout, 256), 256, 0, stream>>>((unsigned*)G,
                                                                   (long)dout * dout);
      colsum_k<<<512, dout, 0, stream>>>(hdst, cs, dout, NN);
      gram_wmma<<<dim3(dout / 16, dout / 16, NN / 160), 32, 0, stream>>>(h16, G, dout, 160);
      corr_finalize<<<1, 256, 0, stream>>>(G, cs, dout, (float)NN,
                                           outf + (size_t)NN * CC + (l == 3 ? 1 : 0));
      if (l == 2)  // next layer's GEMM input is ReLU(h); relu(f16) == f16(relu)
        relu_f16<<<blocks((long)NN * dout, 256), 256, 0, stream>>>(h16, (long)NN * dout);
    } else {
      // ReLU fused into the f16 conversion for the next layer
      f32_to_f16<<<blocks((long)NN * dout, 256), 256, 0, stream>>>(hdst, h16,
                                                                   (long)NN * dout, 1);
    }
  }
}
